// IntraZPConv_39213051412497
// MI455X (gfx1250) — matrix operand order repaired and checked by reading, not compiled, not run on hardware
//
#include <hip/hip_runtime.h>

// ---------------- problem constants (from reference) ----------------
#define BS      2
#define C_IN    64
#define NP      1024
#define NA      60
#define NA_OUT  60
#define KS      3
#define ANN     8
#define C_OUT   128
#define K1      (C_IN * KS)        // 192 : K of GEMM2, N(padded) of GEMM1
#define NA_PAD  64                 // gather K padded to 64
#define O_PAD   64                 // output-angle dim padded to 64

// LDS strides (in 16-bit halves); even (b32-aligned) and bank-friendly
#define SROW 66     // Scm column stride  (64 + 2)
#define FROW 66     // Frm row stride     (64 + 2)
#define WROW 194    // Wrm row stride     (192 + 2)
#define XROW 194    // Xcm column stride  (192 + 2)

#define SCM_HALVES (K1 * SROW)        // 192*66  = 12672
#define WRM_HALVES (C_OUT * WROW)     // 128*194 = 24832
#define FRM_HALVES (C_IN * FROW)      // 64*66   = 4224
#define XCM_HALVES (O_PAD * XROW)     // 64*194  = 12416
#define LDS_HALVES (SCM_HALVES + WRM_HALVES + FRM_HALVES + XCM_HALVES)

#define PTS_PER_WG 4
#define NBLOCKS    ((BS * NP) / PTS_PER_WG)   // 512

typedef __attribute__((ext_vector_type(16))) __bf16 v16bf;
typedef __attribute__((ext_vector_type(2)))  __bf16 bf16x2;
typedef __attribute__((ext_vector_type(8)))  float  v8f;

union Frag { v16bf v; unsigned int u[8]; };
union AccU { v8f  v; float f[8]; };

static __device__ __forceinline__ unsigned short f2bf(float f) {
  __bf16 h = (__bf16)f;
  return __builtin_bit_cast(unsigned short, h);
}
static __device__ __forceinline__ unsigned int pk2(float a, float b) {
  // packs to one v_cvt_pk_bf16_f32
  bf16x2 t; t.x = (__bf16)a; t.y = (__bf16)b;
  return __builtin_bit_cast(unsigned int, t);
}

__global__ __launch_bounds__(256)
void IntraZPConv_39213051412497_kernel(const float* __restrict__ feats,
                                       const float* __restrict__ intra_w,
                                       const float* __restrict__ Wmat,
                                       const float* __restrict__ bias,
                                       const int*   __restrict__ intra_idx,
                                       float* __restrict__ out)
{
  extern __shared__ unsigned short smem[];
  unsigned short* Scm = smem;                       // [n1 = k*64+o][na]     col-major S (bf16)
  unsigned short* Wrm = Scm + SCM_HALVES;           // [m][i' = k*64+c]      row-major W, K-permuted
  unsigned short* Frm = Wrm + WRM_HALVES;           // [c][na]               row-major feats tile
  unsigned short* Xcm = Frm + FRM_HALVES;           // [o][i' = k*64+c]      col-major X (bf16)

  const int tid   = threadIdx.x;
  const int lane  = tid & 31;
  const int wv    = tid >> 5;          // wave 0..7
  const int lmod  = lane & 15;
  const int lhalf = (lane < 16) ? 0 : 1;

  // ---------------- once per block: build S (bf16) and W (bf16) ----------------
  {
    unsigned int* pz = (unsigned int*)Scm;          // zero pad rows/cols of S
    for (int e = tid; e < SCM_HALVES / 2; e += 256) pz[e] = 0u;
  }
  __syncthreads();
  if (tid < NA_OUT * KS) {                          // 180 columns of S
    const int o = tid % NA_OUT, k = tid / NA_OUT;
    const int base = (o * KS + k) * ANN;
    int   idxs[ANN]; float ws[ANN];
    for (int a = 0; a < ANN; ++a) { idxs[a] = intra_idx[base + a]; ws[a] = intra_w[base + a]; }
    unsigned int* colU = (unsigned int*)(Scm + (k * O_PAD + o) * SROW);
    for (int na2 = 0; na2 < NA / 2; ++na2) {        // packed pair stores
      float s0 = 0.f, s1 = 0.f;
      for (int a = 0; a < ANN; ++a) {
        s0 += (idxs[a] == 2 * na2)     ? ws[a] : 0.f;
        s1 += (idxs[a] == 2 * na2 + 1) ? ws[a] : 0.f;
      }
      colU[na2] = pk2(s0, s1);
    }
  }
  // W fp32 -> bf16 LDS with K permutation i' = k*64 + c  (source col = c*3 + k)
  for (int e = tid; e < (C_OUT * K1) / 2; e += 256) {
    const int h = e << 1;
    const int m = h / K1, i = h % K1;               // i even -> c even, pair stays in same k
    const int k = i >> 6, c = i & 63;
    const float a = Wmat[m * K1 + c * KS + k];
    const float bb = Wmat[m * K1 + (c + 1) * KS + k];
    ((unsigned int*)(Wrm + m * WROW))[i >> 1] = pk2(a, bb);
  }
  __syncthreads();

  // ---------------- hoisted per-wave invariants ----------------
  // GEMM2: wave wv owns M-tile wv; its 6 W A-fragments never change -> keep in VGPRs
  Frag WA[6];
  {
    const int m = (wv << 4) + lmod;
    const unsigned int* wrow = (const unsigned int*)(Wrm + m * WROW);
    #pragma unroll
    for (int ks = 0; ks < 6; ++ks) {
      const int k0 = (ks << 5) + (lhalf ? 16 : 0);
      const int k1 = (ks << 5) + (lhalf ? 24 : 8);
      #pragma unroll
      for (int q = 0; q < 4; ++q) {
        WA[ks].u[q]     = wrow[(k0 >> 1) + q];
        WA[ks].u[4 + q] = wrow[(k1 >> 1) + q];
      }
    }
  }
  // bias values for this lane's 8 output rows
  const int mbase = (wv << 4) + (lhalf ? 8 : 0);
  float biasr[8];
  #pragma unroll
  for (int r = 0; r < 8; ++r) biasr[r] = bias[mbase + r];

  // GEMM1: with t = wv + 8*t6, mt = t&3 = wv&3 (constant), nt = (wv>>2) + 2*t6
  const int mt1   = wv & 3;
  const int crow  = (mt1 << 4) + lmod;
  const int cbase = (mt1 << 4) + (lhalf ? 8 : 0);

  // ---------------- per-point pipeline ----------------
  const int flat0 = blockIdx.x * PTS_PER_WG;
  for (int j = 0; j < PTS_PER_WG; ++j) {
    const int flat = flat0 + j;
    const int b = flat >> 10;          // / NP
    const int p = flat & (NP - 1);

    // load feats[b,:,p,:] -> bf16 LDS tile (packed pair stores), pad na 60->64
    const float* fb = feats + ((size_t)(b * C_IN) * NP + p) * NA;
    for (int e = tid; e < (C_IN * NA_PAD) / 2; e += 256) {
      const int h = e << 1;
      const int c = h >> 6, na = h & 63;             // na even; 60-boundary not straddled
      float a = 0.f, bb = 0.f;
      if (na < NA) {
        const float* src = fb + (size_t)c * NP * NA + na;
        a = src[0]; bb = src[1];
      }
      ((unsigned int*)(Frm + c * FROW))[na >> 1] = pk2(a, bb);
    }
    // prefetch next point's feats rows (gfx1250 global_prefetch path)
    if (j + 1 < PTS_PER_WG && tid < C_IN) {
      const int f2 = flat + 1;
      const float* fn = feats + ((size_t)((f2 >> 10) * C_IN) * NP + (f2 & (NP - 1))) * NA;
      __builtin_prefetch(fn + (size_t)tid * NP * NA, 0, 0);
    }
    __syncthreads();

    // ---- GEMM1: Y(64x192) = F(64x64pad) @ S(64x192), packed store into Xcm ----
    {
      // this wave's two F A-fragments (ks = 0,1), reused for all 6 tiles
      Frag FA[2];
      const unsigned int* rowU = (const unsigned int*)(Frm + crow * FROW);
      #pragma unroll
      for (int ks = 0; ks < 2; ++ks) {
        const int k0 = (ks << 5) + (lhalf ? 16 : 0);
        const int k1 = (ks << 5) + (lhalf ? 24 : 8);
        #pragma unroll
        for (int q = 0; q < 4; ++q) {
          FA[ks].u[q]     = rowU[(k0 >> 1) + q];
          FA[ks].u[4 + q] = rowU[(k1 >> 1) + q];
        }
      }
      for (int t6 = 0; t6 < 6; ++t6) {
        const int nt = (wv >> 2) + (t6 << 1);     // 0..11
        const int n1 = (nt << 4) + lmod;
        AccU acc; acc.v = (v8f){0.f,0.f,0.f,0.f,0.f,0.f,0.f,0.f};
        #pragma unroll
        for (int ks = 0; ks < 2; ++ks) {
          Frag B;
          const unsigned int* colU = (const unsigned int*)(Scm + n1 * SROW);
          const int kb = (ks << 5) + (lhalf ? 16 : 0);
          #pragma unroll
          for (int q = 0; q < 8; ++q) B.u[q] = colU[(kb >> 1) + q];
          acc.v = __builtin_amdgcn_wmma_f32_16x16x32_bf16(false, FA[ks].v, false, B.v,
                                                          (short)0, acc.v, false, false);
        }
        // D rows c = cbase+r are CONSECUTIVE halves of X column o at offset k*64+c
        const int k = n1 >> 6, o = n1 & 63;            // n1 = k*64 + o
        unsigned int* xcU = (unsigned int*)(Xcm + o * XROW + (k << 6) + cbase);
        #pragma unroll
        for (int r2 = 0; r2 < 4; ++r2)
          xcU[r2] = pk2(acc.f[2 * r2], acc.f[2 * r2 + 1]);
      }
    }
    __syncthreads();

    // ---- GEMM2: out(128x60) = W(128x192) @ X(192x64pad); A-frags resident ----
    {
      AccU acc[4];
      #pragma unroll
      for (int nt = 0; nt < 4; ++nt) acc[nt].v = (v8f){0.f,0.f,0.f,0.f,0.f,0.f,0.f,0.f};
      #pragma unroll
      for (int ks = 0; ks < 6; ++ks) {
        const int kb = (ks << 5) + (lhalf ? 16 : 0);
        #pragma unroll
        for (int nt = 0; nt < 4; ++nt) {
          Frag B;
          const unsigned int* colU = (const unsigned int*)(Xcm + ((nt << 4) + lmod) * XROW);
          #pragma unroll
          for (int q = 0; q < 8; ++q) B.u[q] = colU[(kb >> 1) + q];
          acc[nt].v = __builtin_amdgcn_wmma_f32_16x16x32_bf16(false, WA[ks].v, false, B.v,
                                                              (short)0, acc[nt].v, false, false);
        }
      }
      // bias + store (drop padded cols o>=60)
      #pragma unroll
      for (int nt = 0; nt < 4; ++nt) {
        const int o = (nt << 4) + lmod;
        if (o < NA_OUT) {
          #pragma unroll
          for (int r = 0; r < 8; ++r) {
            const int mm = mbase + r;
            out[((size_t)(b * C_OUT + mm) * NP + p) * NA_OUT + o] = acc[nt].f[r] + biasr[r];
          }
        }
      }
    }
    __syncthreads();   // before Frm/Xcm are overwritten for next point
  }
}

extern "C" void kernel_launch(void* const* d_in, const int* in_sizes, int n_in,
                              void* d_out, int out_size, void* d_ws, size_t ws_size,
                              hipStream_t stream) {
  const float* feats     = (const float*)d_in[0];
  const float* intra_w   = (const float*)d_in[1];
  const float* Wmat      = (const float*)d_in[2];
  const float* bias      = (const float*)d_in[3];
  const int*   intra_idx = (const int*)  d_in[4];
  float* out = (float*)d_out;

  const size_t lds_bytes = (size_t)LDS_HALVES * sizeof(unsigned short); // ~106 KB of the 320 KB WGP LDS
  IntraZPConv_39213051412497_kernel<<<dim3(NBLOCKS), dim3(256), lds_bytes, stream>>>(
      feats, intra_w, Wmat, bias, intra_idx, out);
}
